// QuantumAttentionMechanism_71854802862781
// MI455X (gfx1250) — matrix operand order repaired and checked by reading, not compiled, not run on hardware
//
#include <hip/hip_runtime.h>
#include <hip/hip_bf16.h>
#include <math.h>

// ---------------------------------------------------------------------------
// QuantumAttentionMechanism for MI455X (gfx1250).
// Heavy GEMMs + flash attention run on v_wmma_f32_16x16x32_f16 with
// async-to-LDS double-buffered tiles (ASYNCcnt pipeline).
// ---------------------------------------------------------------------------

typedef _Float16 half_t;
typedef __attribute__((ext_vector_type(8)))  half_t v8h;
typedef __attribute__((ext_vector_type(16))) half_t v16h;
typedef __attribute__((ext_vector_type(8)))  float  v8f;

#define BB 8
#define NN 1024
#define CC 768
#define HH 8
#define DD 96
#define QQ 8

// ---------------- WMMA fragment helpers (wave32, ISA 7.12.2 layouts) -------

__device__ inline v16h load_frag_a(const half_t* __restrict__ base, int ld) {
    // 16x32 f16 A tile, row-major with leading dim ld.
    int lane = threadIdx.x & 31;
    int row  = lane & 15;
    int kb   = (lane >> 4) << 3;     // 0 or 8
    const half_t* p = base + (size_t)row * ld + kb;
    union { v16h v; v8h h[2]; } u;
    u.h[0] = *(const v8h*)(p);
    u.h[1] = *(const v8h*)(p + 16);
    return u.v;
}

__device__ inline v16h load_frag_b(const half_t* __restrict__ base, int ld) {
    // 32x16 f16 B tile, supplied as B^T (N x K) row-major with leading dim ld.
    int lane = threadIdx.x & 31;
    int col  = lane & 15;
    int kb   = (lane >> 4) << 3;     // 0 or 8
    const half_t* p = base + (size_t)col * ld + kb;
    union { v16h v; v8h h[2]; } u;
    u.h[0] = *(const v8h*)(p);
    u.h[1] = *(const v8h*)(p + 16);
    return u.v;
}

__device__ inline v8f wmma32(v16h a, v16h b, v8f c) {
    return __builtin_amdgcn_wmma_f32_16x16x32_f16(false, a, false, b, (short)0, c,
                                                  false, false);
}

__device__ inline uint32_t lds_offset_of(const half_t* p) {
    // generic -> addrspace(3) cast yields the 32-bit LDS offset
    return (uint32_t)(uintptr_t)(__attribute__((address_space(3))) const half_t*)p;
}

__device__ inline void async_load_b128_to_lds(const half_t* gsrc, const half_t* lds_dst) {
    uint32_t loff = lds_offset_of(lds_dst);
    asm volatile("global_load_async_to_lds_b128 %0, %1, off"
                 :
                 : "v"(loff), "v"((uint64_t)(uintptr_t)gsrc)
                 : "memory");
}

__device__ inline void wait_asynccnt0() {
    asm volatile("s_wait_asynccnt 0x0" ::: "memory");
}

__device__ inline float half16_reduce_max(float v) {
    v = fmaxf(v, __shfl_xor(v, 1, 32));
    v = fmaxf(v, __shfl_xor(v, 2, 32));
    v = fmaxf(v, __shfl_xor(v, 4, 32));
    v = fmaxf(v, __shfl_xor(v, 8, 32));
    return v;
}
__device__ inline float half16_reduce_sum(float v) {
    v += __shfl_xor(v, 1, 32);
    v += __shfl_xor(v, 2, 32);
    v += __shfl_xor(v, 4, 32);
    v += __shfl_xor(v, 8, 32);
    return v;
}

// ---------------- pack / convert kernels -----------------------------------

__global__ void f32_to_f16_kernel(const float* __restrict__ src,
                                  half_t* __restrict__ dst, size_t n) {
    size_t i = (size_t)blockIdx.x * blockDim.x + threadIdx.x;
    if (i < n) dst[i] = (half_t)src[i];
}

__global__ void f32_to_f16_T_kernel(const float* __restrict__ src,
                                    half_t* __restrict__ dst, int rows, int cols) {
    int c = blockIdx.x * blockDim.x + threadIdx.x;
    int r = blockIdx.y * blockDim.y + threadIdx.y;
    if (c < cols && r < rows)
        dst[(size_t)c * rows + r] = (half_t)src[(size_t)r * cols + c];
}

// qkv (B,N,3,H,D) f32 -> qh,kh (B,H,N,D) f16 and vT (B,H,D,N) f16
__global__ void split_pack_kernel(const float* __restrict__ qkv,
                                  half_t* __restrict__ qh, half_t* __restrict__ kh,
                                  half_t* __restrict__ vT) {
    size_t i = (size_t)blockIdx.x * blockDim.x + threadIdx.x;
    const size_t total = (size_t)BB * HH * NN * DD;
    if (i >= total) return;
    int d = i % DD; size_t t = i / DD;
    int n = t % NN; t /= NN;
    int h = t % HH; int b = t / HH;
    size_t base = (((size_t)(b * NN + n) * 3) * HH + h) * DD + d;
    float qv = qkv[base];
    float kv = qkv[base + (size_t)HH * DD];
    float vv = qkv[base + (size_t)2 * HH * DD];
    size_t hd = ((size_t)(b * HH + h) * NN + n) * DD + d;
    qh[hd] = (half_t)qv;
    kh[hd] = (half_t)kv;
    vT[((size_t)(b * HH + h) * DD + d) * NN + n] = (half_t)vv;
}

// ---------------- WMMA GEMM: C(MxN) = A(MxK) @ Bt(NxK)^T + bias (+ residual)
// Block = 8 waves = 128(M) x 64(N) tile. The shared 64x32 B slice streams into
// LDS via global_load_async_to_lds_b128, double-buffered; the private A
// fragment is software-pipelined in registers.

__global__ __launch_bounds__(256) void wmma_gemm_kernel(
    const half_t* __restrict__ A, const half_t* __restrict__ Bt,
    const float* __restrict__ bias, const float* __restrict__ residual,
    float* __restrict__ C, int M, int N, int K) {
    __shared__ half_t sB[2][64 * 32];           // two 4KB B-tile buffers
    const int tid  = threadIdx.x;
    const int wave = tid >> 5;
    const int lane = tid & 31;
    const int m0 = blockIdx.y * 128 + wave * 16;
    const int n0 = blockIdx.x * 64;

    const int brow = tid >> 2;                  // 0..63 : B^T row (n within tile)
    const int bch  = (tid & 3) << 3;            // 0,8,16,24 : k chunk (8 halfs)

    auto stage_B = [&](int buf, int k0) {
        const half_t* g = Bt + (size_t)(n0 + brow) * K + k0 + bch;
        async_load_b128_to_lds(g, &sB[buf][brow * 32 + bch]);
    };

    stage_B(0, 0);
    v16h a = load_frag_a(A + (size_t)m0 * K, K);
    v8f acc[4] = {v8f{}, v8f{}, v8f{}, v8f{}};
    int cur = 0;
    for (int k0 = 0; k0 < K; k0 += 32) {
        int knext = (k0 + 32 < K) ? k0 + 32 : 0;   // last-iter stage is dead data
        wait_asynccnt0();            // my slice of sB[cur] has landed
        __syncthreads();             // everyone's slice landed; prev reads done
        stage_B(cur ^ 1, knext);                   // prefetch next slice
        v16h a_next = load_frag_a(A + (size_t)m0 * K + knext, K);
        v16h bf[4];
#pragma unroll
        for (int t = 0; t < 4; ++t) bf[t] = load_frag_b(&sB[cur][16 * t * 32], 32);
#pragma unroll
        for (int t = 0; t < 4; ++t) acc[t] = wmma32(a, bf[t], acc[t]);
        a = a_next;
        cur ^= 1;
    }
    const int cn = lane & 15;
    const int mb = (lane >> 4) << 3;
#pragma unroll
    for (int t = 0; t < 4; ++t) {
        int nn = n0 + 16 * t + cn;
        float bv = bias ? bias[nn] : 0.0f;
#pragma unroll
        for (int r = 0; r < 8; ++r) {
            int mm = m0 + mb + r;
            float val = acc[t][r] + bv;
            if (residual) val += residual[(size_t)mm * N + nn];
            C[(size_t)mm * N + nn] = val;
        }
    }
}

// ---------------- flash attention (classical path) --------------------------
// Block = 8 waves, one head. K-tile (32x96) and V^T-tile (96x32) are shared by
// all waves -> async-to-LDS double-buffered. Q A-fragments are loop-invariant
// and live in registers. One wave owns 16 query rows; online softmax.

__global__ __launch_bounds__(256) void flash_attn_kernel(
    const half_t* __restrict__ qh, const half_t* __restrict__ kh,
    const half_t* __restrict__ vT, float* __restrict__ out, float scale) {
    __shared__ half_t sK[2][32 * 96];      // 6KB x2 : keys x d
    __shared__ half_t sV[2][96 * 32];      // 6KB x2 : d x keys
    __shared__ half_t sP[8][16 * 32];      // per-wave P staging
    const int tid  = threadIdx.x;
    const int wave = tid >> 5;
    const int lane = tid & 31;
    const int bh = blockIdx.x;             // b*H + h
    const int m0 = (blockIdx.y * 8 + wave) * 16;
    const int b = bh / HH, h = bh % HH;

    const half_t* qhead = qh + (size_t)bh * NN * DD;
    const half_t* khead = kh + (size_t)bh * NN * DD;
    const half_t* vthead = vT + (size_t)bh * DD * NN;

    const int cn = lane & 15;
    const int mb = (lane >> 4) << 3;

    // Stage K tile (32 rows x 96 halfs = 384 b128 chunks) and V^T tile
    // (96 rows x 32 halfs = 384 chunks): 3 async issues per thread.
    auto stage_tiles = [&](int buf, int j0) {
#pragma unroll
        for (int r = 0; r < 3; ++r) {
            int c = tid + r * 256;                       // 0..767
            if (c < 384) {
                int row = c / 12, ch = (c % 12) * 8;
                async_load_b128_to_lds(khead + (size_t)(j0 + row) * DD + ch,
                                       &sK[buf][row * 96 + ch]);
            } else {
                int c2 = c - 384;
                int row = c2 / 4, ch = (c2 % 4) * 8;
                async_load_b128_to_lds(vthead + (size_t)row * NN + j0 + ch,
                                       &sV[buf][row * 32 + ch]);
            }
        }
    };

    // loop-invariant Q fragments (16 rows x 96 over three k-slices)
    v16h aq[3];
#pragma unroll
    for (int kk = 0; kk < 3; ++kk)
        aq[kk] = load_frag_a(qhead + (size_t)m0 * DD + 32 * kk, DD);

    v8f accO[6] = {v8f{}, v8f{}, v8f{}, v8f{}, v8f{}, v8f{}};
    float m_row[8], l_row[8];
#pragma unroll
    for (int r = 0; r < 8; ++r) { m_row[r] = -1e30f; l_row[r] = 0.0f; }

    stage_tiles(0, 0);
    int cur = 0;
    for (int j0 = 0; j0 < NN; j0 += 32) {
        int jnext = (j0 + 32 < NN) ? j0 + 32 : 0;  // last-iter stage is dead data
        wait_asynccnt0();
        __syncthreads();
        stage_tiles(cur ^ 1, jnext);

        // S tiles (16x16 each) over the D=96 contraction, fragments up front
        v16h bk[6];
#pragma unroll
        for (int kk = 0; kk < 3; ++kk) {
            bk[kk * 2 + 0] = load_frag_b(&sK[cur][0 * 96 + 32 * kk], 96);
            bk[kk * 2 + 1] = load_frag_b(&sK[cur][16 * 96 + 32 * kk], 96);
        }
        v8f s0 = v8f{}, s1 = v8f{};
#pragma unroll
        for (int kk = 0; kk < 3; ++kk) {
            s0 = wmma32(aq[kk], bk[kk * 2 + 0], s0);
            s1 = wmma32(aq[kk], bk[kk * 2 + 1], s1);
        }
        // online softmax per row (rows live across the 16 lanes of each half)
#pragma unroll
        for (int r = 0; r < 8; ++r) {
            float v0 = s0[r] * scale;
            float v1 = s1[r] * scale;
            float mx = half16_reduce_max(fmaxf(v0, v1));
            float newm = fmaxf(m_row[r], mx);
            float factor = __expf(m_row[r] - newm);
            float p0 = __expf(v0 - newm);
            float p1 = __expf(v1 - newm);
            float rs = half16_reduce_sum(p0 + p1);
            l_row[r] = l_row[r] * factor + rs;
            m_row[r] = newm;
#pragma unroll
            for (int t = 0; t < 6; ++t) accO[t][r] *= factor;
            sP[wave][(mb + r) * 32 + cn]      = (half_t)p0;
            sP[wave][(mb + r) * 32 + 16 + cn] = (half_t)p1;
        }
        // O += P @ V  (A = P 16x32 from LDS, B^T = V^T tile in LDS)
        v16h ap = load_frag_a(&sP[wave][0], 32);
        v16h bv[6];
#pragma unroll
        for (int t = 0; t < 6; ++t) bv[t] = load_frag_b(&sV[cur][16 * t * 32], 32);
#pragma unroll
        for (int t = 0; t < 6; ++t) accO[t] = wmma32(ap, bv[t], accO[t]);
        cur ^= 1;
    }
    // normalize + store into (B,N,C)
#pragma unroll
    for (int r = 0; r < 8; ++r) {
        float inv_l = 1.0f / l_row[r];
        int row = m0 + mb + r;
        float* op = out + ((size_t)b * NN + row) * CC + h * DD;
#pragma unroll
        for (int t = 0; t < 6; ++t) op[16 * t + cn] = accO[t][r] * inv_l;
    }
}

// ---------------- quantum path ----------------------------------------------

__global__ __launch_bounds__(256) void quantum_kernel(
    const float* __restrict__ qkv,
    const float* __restrict__ qe_w1, const float* __restrict__ qe_b1,
    const float* __restrict__ qe_w2, const float* __restrict__ qe_b2,
    const float* __restrict__ ke_w1, const float* __restrict__ ke_b1,
    const float* __restrict__ ke_w2, const float* __restrict__ ke_b2,
    const float* __restrict__ gate, const float* __restrict__ ent,
    const float* __restrict__ md_w1, const float* __restrict__ md_b1,
    const float* __restrict__ md_w2, const float* __restrict__ md_b2,
    const float* __restrict__ gumbel, float* __restrict__ qo) {
    int i = blockIdx.x * blockDim.x + threadIdx.x;
    if (i >= BB * HH * NN) return;
    int n = i % NN; int t = i / NN;
    int h = t % HH; int b = t / HH;

    const float* qrow = qkv + (((size_t)(b * NN + n) * 3 + 0) * HH + h) * DD;
    const float* krow = qkv + (((size_t)(b * NN + n) * 3 + 1) * HH + h) * DD;
    const float* vrow = qkv + (((size_t)(b * NN + n) * 3 + 2) * HH + h) * DD;

    float sre[QQ], sim[QQ];
#pragma unroll
    for (int q = 0; q < QQ; ++q) { sre[q] = 0.0f; sim[q] = 0.0f; }

    // two embeds (q and k); v-embed in reference is dead code
    for (int which = 0; which < 2; ++which) {
        const float* row = which ? krow : qrow;
        const float* w1 = which ? ke_w1 : qe_w1;
        const float* b1 = which ? ke_b1 : qe_b1;
        const float* w2 = which ? ke_w2 : qe_w2;
        const float* b2 = which ? ke_b2 : qe_b2;
        float hid[64];
#pragma unroll
        for (int j = 0; j < 64; ++j) hid[j] = b1[j];
        for (int d = 0; d < DD; ++d) {
            float xv = row[d];
            const float* wr = w1 + d * 64;
#pragma unroll
            for (int j = 0; j < 64; ++j) hid[j] += xv * wr[j];
        }
#pragma unroll
        for (int j = 0; j < 64; ++j) hid[j] = fmaxf(hid[j], 0.0f);
        float e[16];
#pragma unroll
        for (int j = 0; j < 16; ++j) e[j] = b2[j];
        for (int d = 0; d < 64; ++d) {
            float xv = hid[d];
            const float* wr = w2 + d * 16;
#pragma unroll
            for (int j = 0; j < 16; ++j) e[j] += xv * wr[j];
        }
        float amax = -1e30f;
#pragma unroll
        for (int q = 0; q < QQ; ++q) amax = fmaxf(amax, e[q]);
        float asum = 0.0f;
        float amp[QQ];
#pragma unroll
        for (int q = 0; q < QQ; ++q) { amp[q] = __expf(e[q] - amax); asum += amp[q]; }
        float inv = 1.0f / asum;
#pragma unroll
        for (int q = 0; q < QQ; ++q) {
            float a = amp[q] * inv;
            float ph = e[QQ + q];
            sre[q] += a * __cosf(ph);
            sim[q] += a * __sinf(ph);
        }
    }

    // variational layers (depth 2)
#pragma unroll
    for (int layer = 0; layer < 2; ++layer) {
#pragma unroll
        for (int q = 0; q < QQ; ++q) {
            float ry = gate[(layer * QQ + q) * 3 + 1];
            float rz = gate[(layer * QQ + q) * 3 + 2];
            float cz = __cosf(rz), sz = __sinf(rz);
            float nr = sre[q] * cz - sim[q] * sz;
            float ni = sre[q] * sz + sim[q] * cz;
            float f = __cosf(ry * 0.5f);
            sre[q] = nr * f;
            sim[q] = ni * f;
        }
        float pre[QQ], pim[QQ];
#pragma unroll
        for (int q = 0; q < QQ; ++q) { pre[q] = sre[q]; pim[q] = sim[q]; }
#pragma unroll
        for (int q = 1; q < QQ; ++q) {
            float s = 1.0f / (1.0f + __expf(-ent[layer * (QQ - 1) + (q - 1)]));
            sre[q] = pre[q] + s * pre[q - 1];
            sim[q] = pim[q] + s * pim[q - 1];
        }
    }

    // probs = softmax(|state|^2), meas = softmax((log(probs+eps)+gumbel)/0.5)
    float p2[QQ];
    float pmax = -1e30f;
#pragma unroll
    for (int q = 0; q < QQ; ++q) {
        p2[q] = sre[q] * sre[q] + sim[q] * sim[q];
        pmax = fmaxf(pmax, p2[q]);
    }
    float psum = 0.0f;
#pragma unroll
    for (int q = 0; q < QQ; ++q) { p2[q] = __expf(p2[q] - pmax); psum += p2[q]; }
    float pinv = 1.0f / psum;
    const float* grow = gumbel + ((size_t)(b * HH + h) * NN + n) * QQ;
    float lg[QQ];
    float lmax = -1e30f;
#pragma unroll
    for (int q = 0; q < QQ; ++q) {
        lg[q] = (__logf(p2[q] * pinv + 1e-10f) + grow[q]) * 2.0f;
        lmax = fmaxf(lmax, lg[q]);
    }
    float lsum = 0.0f;
#pragma unroll
    for (int q = 0; q < QQ; ++q) { lg[q] = __expf(lg[q] - lmax); lsum += lg[q]; }
    float linv = 1.0f / lsum;

    // measurement decoder MLP: 8 -> 64 (relu) -> 96
    float hid2[64];
#pragma unroll
    for (int j = 0; j < 64; ++j) hid2[j] = md_b1[j];
#pragma unroll
    for (int q = 0; q < QQ; ++q) {
        float m = lg[q] * linv;
        const float* wr = md_w1 + q * 64;
#pragma unroll
        for (int j = 0; j < 64; ++j) hid2[j] += m * wr[j];
    }
#pragma unroll
    for (int j = 0; j < 64; ++j) hid2[j] = fmaxf(hid2[j], 0.0f);

    float* op = qo + ((size_t)b * NN + n) * CC + h * DD;
    for (int d = 0; d < DD; ++d) {
        float s = md_b2[d];
#pragma unroll
        for (int j = 0; j < 64; ++j) s += hid2[j] * md_w2[j * DD + d];
        op[d] = s * vrow[d];
    }
}

// ---------------- small glue kernels ----------------------------------------

__global__ void xmean_kernel(const float* __restrict__ x, float* __restrict__ xm) {
    int i = blockIdx.x * blockDim.x + threadIdx.x;
    if (i >= BB * CC) return;
    int c = i % CC, b = i / CC;
    float s = 0.0f;
    for (int n = 0; n < NN; ++n) s += x[((size_t)b * NN + n) * CC + c];
    xm[i] = s * (1.0f / NN);
}

__global__ void fw_kernel(const float* __restrict__ xm, const float* __restrict__ w_proj,
                          const float* __restrict__ b_proj, float* __restrict__ fw) {
    int i = blockIdx.x * blockDim.x + threadIdx.x;
    if (i >= BB * CC) return;
    int c = i % CC, b = i / CC;
    float s = b_proj[c];
    for (int k = 0; k < CC; ++k) s += xm[b * CC + k] * w_proj[(size_t)k * CC + c];
    fw[i] = 1.0f / (1.0f + __expf(-s));
}

__global__ void fuse_kernel(const float* __restrict__ fw, const float* __restrict__ qo,
                            const float* __restrict__ co, half_t* __restrict__ fusedh) {
    size_t i = (size_t)blockIdx.x * blockDim.x + threadIdx.x;
    const size_t total = (size_t)BB * NN * CC;
    if (i >= total) return;
    int c = i % CC;
    int b = i / ((size_t)NN * CC);
    float f = fw[b * CC + c];
    fusedh[i] = (half_t)(f * qo[i] + (1.0f - f) * co[i]);
}

__global__ __launch_bounds__(256) void layernorm_kernel(
    const float* __restrict__ y, const float* __restrict__ gamma,
    const float* __restrict__ beta, float* __restrict__ out) {
    int row = blockIdx.x * 8 + (threadIdx.x >> 5);
    int lane = threadIdx.x & 31;
    const float* yr = y + (size_t)row * CC;
    float s = 0.0f;
    for (int c = lane; c < CC; c += 32) s += yr[c];
#pragma unroll
    for (int off = 1; off < 32; off <<= 1) s += __shfl_xor(s, off, 32);
    float mu = s * (1.0f / CC);
    float v = 0.0f;
    for (int c = lane; c < CC; c += 32) { float d = yr[c] - mu; v += d * d; }
#pragma unroll
    for (int off = 1; off < 32; off <<= 1) v += __shfl_xor(v, off, 32);
    float inv = rsqrtf(v * (1.0f / CC) + 1e-5f);
    float* op = out + (size_t)row * CC;
    for (int c = lane; c < CC; c += 32) op[c] = (yr[c] - mu) * inv * gamma[c] + beta[c];
}

// ---------------- launch ----------------------------------------------------

extern "C" void kernel_launch(void* const* d_in, const int* in_sizes, int n_in,
                              void* d_out, int out_size, void* d_ws, size_t ws_size,
                              hipStream_t stream) {
    const float* x      = (const float*)d_in[0];
    const float* w_qkv  = (const float*)d_in[1];
    const float* b_qkv  = (const float*)d_in[2];
    const float* w_proj = (const float*)d_in[3];
    const float* b_proj = (const float*)d_in[4];
    const float* gamma  = (const float*)d_in[5];
    const float* beta   = (const float*)d_in[6];
    const float* qe_w1  = (const float*)d_in[7];
    const float* qe_b1  = (const float*)d_in[8];
    const float* qe_w2  = (const float*)d_in[9];
    const float* qe_b2  = (const float*)d_in[10];
    const float* ke_w1  = (const float*)d_in[11];
    const float* ke_b1  = (const float*)d_in[12];
    const float* ke_w2  = (const float*)d_in[13];
    const float* ke_b2  = (const float*)d_in[14];
    // d_in[15..18] = ve_* : dead code in the reference (result discarded)
    const float* gate   = (const float*)d_in[19];
    const float* ent    = (const float*)d_in[20];
    const float* md_w1  = (const float*)d_in[21];
    const float* md_b1  = (const float*)d_in[22];
    const float* md_w2  = (const float*)d_in[23];
    const float* md_b2  = (const float*)d_in[24];
    const float* gumbel = (const float*)d_in[25];
    float* out = (float*)d_out;

    char* w = (char*)d_ws;
    size_t off = 0;
    auto carve = [&](size_t bytes) -> void* {
        off = (off + 255) & ~(size_t)255;
        void* p = w + off;
        off += bytes;
        return p;
    };
    const size_t MN = (size_t)BB * NN;          // 8192
    half_t* xh     = (half_t*)carve(MN * CC * 2);
    half_t* wqkvT  = (half_t*)carve((size_t)3 * CC * CC * 2);
    half_t* wprojT = (half_t*)carve((size_t)CC * CC * 2);
    float*  qkvbuf = (float*) carve(MN * 3 * CC * 4);
    half_t* qh     = (half_t*)carve((size_t)BB * HH * NN * DD * 2);
    half_t* kh     = (half_t*)carve((size_t)BB * HH * NN * DD * 2);
    half_t* vT     = (half_t*)carve((size_t)BB * HH * DD * NN * 2);
    float*  co     = (float*) carve(MN * CC * 4);
    float*  qo     = (float*) carve(MN * CC * 4);
    float*  xm     = (float*) carve((size_t)BB * CC * 4);
    float*  fw     = (float*) carve((size_t)BB * CC * 4);
    half_t* fusedh = (half_t*)carve(MN * CC * 2);
    float*  ybuf   = (float*) carve(MN * CC * 4);

    const size_t nxc = MN * CC;                 // 6291456

    // 1. pack x -> f16
    f32_to_f16_kernel<<<(unsigned)((nxc + 255) / 256), 256, 0, stream>>>(x, xh, nxc);
    // 2. pack weights transposed -> f16
    {
        dim3 tb(16, 16);
        f32_to_f16_T_kernel<<<dim3((3 * CC + 15) / 16, (CC + 15) / 16), tb, 0, stream>>>(
            w_qkv, wqkvT, CC, 3 * CC);
        f32_to_f16_T_kernel<<<dim3((CC + 15) / 16, (CC + 15) / 16), tb, 0, stream>>>(
            w_proj, wprojT, CC, CC);
    }
    // 3. QKV GEMM: (8192 x 768) @ (768 x 2304) + b_qkv
    wmma_gemm_kernel<<<dim3(3 * CC / 64, (unsigned)(MN / 128)), 256, 0, stream>>>(
        xh, wqkvT, b_qkv, nullptr, qkvbuf, (int)MN, 3 * CC, CC);
    // 4. split q/k/v into head-major f16 (+ v transposed)
    {
        size_t total = (size_t)BB * HH * NN * DD;
        split_pack_kernel<<<(unsigned)((total + 255) / 256), 256, 0, stream>>>(
            qkvbuf, qh, kh, vT);
    }
    // 5. gating weights
    xmean_kernel<<<(BB * CC + 255) / 256, 256, 0, stream>>>(x, xm);
    fw_kernel<<<(BB * CC + 255) / 256, 256, 0, stream>>>(xm, w_proj, b_proj, fw);
    // 6. classical attention (flash, WMMA) -> co
    {
        float scale = rsqrtf((float)DD);
        flash_attn_kernel<<<dim3(BB * HH, NN / 128), 256, 0, stream>>>(
            qh, kh, vT, co, scale);
    }
    // 7. quantum path -> qo
    quantum_kernel<<<(BB * HH * NN + 255) / 256, 256, 0, stream>>>(
        qkvbuf, qe_w1, qe_b1, qe_w2, qe_b2, ke_w1, ke_b1, ke_w2, ke_b2,
        gate, ent, md_w1, md_b1, md_w2, md_b2, gumbel, qo);
    // 8. fuse -> f16 for projection GEMM
    fuse_kernel<<<(unsigned)((nxc + 255) / 256), 256, 0, stream>>>(fw, qo, co, fusedh);
    // 9. y = x + fused @ w_proj + b_proj
    wmma_gemm_kernel<<<dim3(CC / 64, (unsigned)(MN / 128)), 256, 0, stream>>>(
        fusedh, wprojT, b_proj, x, ybuf, (int)MN, CC, CC);
    // 10. layernorm -> out
    layernorm_kernel<<<(unsigned)(MN / 8), 256, 0, stream>>>(ybuf, gamma, beta, out);
}
